// TFLevitAttention_26379689132085
// MI455X (gfx1250) — matrix operand order repaired and checked by reading, not compiled, not run on hardware
//
#include <hip/hip_runtime.h>

#define S_    196
#define SP    208      // 13*16 padded sequence
#define D_    384
#define HEADS_ 12
#define KD    32
#define VD    64
#define QKV_  1536
#define PIN_  768
#define LDA   40       // bf16 LDS stride for 32-wide K tiles (80B rows, 16B aligned)
#define LDV   232      // vT LDS stride (464B rows)
#define LDP   224      // probs LDS stride (7*32 K padding)
#define LDSC  197      // f32 score LDS stride

typedef __bf16 bf16_t;
typedef __attribute__((ext_vector_type(16))) __bf16 v16bf;
typedef __attribute__((ext_vector_type(8)))  float  v8f;
typedef __attribute__((ext_vector_type(4)))  unsigned int v4u;

union FragBF { v4u u[2]; v16bf v; };

static __device__ __forceinline__ bf16_t f2bf(float x) { return (bf16_t)x; }

static __device__ __forceinline__ v8f wmma_bf16(v16bf a, v16bf b, v8f c) {
  // D = A(16x32 bf16) * B(32x16 bf16) + C(16x16 f32)
  return __builtin_amdgcn_wmma_f32_16x16x32_bf16(false, a, false, b, (short)0, c, false, false);
}

// CDNA5 async global->LDS copy, 16B per lane, tracked by ASYNCcnt.
// LDS offset = low 32 bits of the generic pointer (flat->LDS aperture trunc).
static __device__ __forceinline__ void async_copy16(void* lds_dst, const void* gsrc) {
  unsigned int loff = (unsigned int)(unsigned long long)lds_dst;
  asm volatile("global_load_async_to_lds_b128 %0, %1, off"
               :: "v"(loff), "v"(gsrc) : "memory");
}
static __device__ __forceinline__ void async_wait0() {
  asm volatile("s_wait_asynccnt 0x0" ::: "memory");
}
// allow the 2 newest async ops (next tile's DMA) to stay in flight
static __device__ __forceinline__ void async_wait2() {
  asm volatile("s_wait_asynccnt 0x2" ::: "memory");
}

// ---------------------------------------------------------------------------
// Kernel 0: expand attention bias gather once: biasx[h][sq][sk]
// ---------------------------------------------------------------------------
__global__ void bias_expand_kernel(const float* __restrict__ biases,
                                   const int* __restrict__ idxs,
                                   float* __restrict__ out, int noff) {
  int tid = blockIdx.x * blockDim.x + threadIdx.x;
  const int total = HEADS_ * S_ * S_;
  if (tid >= total) return;
  int h = tid / (S_ * S_);
  int r = tid - h * (S_ * S_);
  out[tid] = biases[h * noff + idxs[r]];
}

// ---------------------------------------------------------------------------
// Kernel 1: qkv = X @ W_qkv, fused BN1, split/convert to bf16 q/k/v buffers
// block: 256 threads (8 waves), tile 128(M) x 128(N), K-steps of 32
// ---------------------------------------------------------------------------
__global__ __launch_bounds__(256) void qkv_gemm_kernel(
    const float* __restrict__ X, const float* __restrict__ W,
    const float* __restrict__ g1, const float* __restrict__ b1,
    const float* __restrict__ m1, const float* __restrict__ v1,
    bf16_t* __restrict__ qb, bf16_t* __restrict__ kb, bf16_t* __restrict__ vb,
    int M) {
  __shared__ bf16_t sA[128 * LDA];   // [m][k]
  __shared__ bf16_t sB[128 * LDA];   // transposed: [n][k]
  const int tid = threadIdx.x;
  const int lane = tid & 31, wave = tid >> 5;
  const int wm = wave & 3, wn = wave >> 2;        // 4 x 2 wave grid
  const int l16 = lane & 15, hh = lane >> 4;
  const int row0 = blockIdx.x * 128;
  const int col0 = blockIdx.y * 128;

  v8f zerov = {};
  v8f acc[2][4];
#pragma unroll
  for (int i = 0; i < 2; ++i)
#pragma unroll
    for (int j = 0; j < 4; ++j) acc[i][j] = zerov;

#pragma unroll
  for (int kt = 0; kt < D_ / 32; ++kt) {
    const int k0 = kt * 32;
    // prefetch next K-step's tiles (WGP scope: locality=3)
    if (k0 + 32 < D_) {
      int rp = row0 + (tid >> 1); if (rp >= M) rp = M - 1;
      __builtin_prefetch(X + (size_t)rp * D_ + k0 + 32, 0, 3);
      __builtin_prefetch(W + (size_t)(k0 + 32 + (tid & 31)) * QKV_ + col0, 0, 3);
    }
    // A tile: 128x32 f32 -> bf16 (1024 float4, 4 per thread)
#pragma unroll
    for (int i = 0; i < 4; ++i) {
      int idx = tid + i * 256;
      int r = idx >> 3, c4 = idx & 7;
      int rr = row0 + r; if (rr >= M) rr = M - 1;
      const float4 x = *(const float4*)(X + (size_t)rr * D_ + k0 + c4 * 4);
      bf16_t* dst = sA + r * LDA + c4 * 4;
      dst[0] = f2bf(x.x); dst[1] = f2bf(x.y); dst[2] = f2bf(x.z); dst[3] = f2bf(x.w);
    }
    // B tile: 32x128 f32 -> bf16, transposed into [n][k]
#pragma unroll
    for (int i = 0; i < 4; ++i) {
      int idx = tid + i * 256;
      int kk = idx >> 5, c4 = idx & 31;
      const float4 w = *(const float4*)(W + (size_t)(k0 + kk) * QKV_ + col0 + c4 * 4);
      int cb = c4 * 4;
      sB[(cb + 0) * LDA + kk] = f2bf(w.x);
      sB[(cb + 1) * LDA + kk] = f2bf(w.y);
      sB[(cb + 2) * LDA + kk] = f2bf(w.z);
      sB[(cb + 3) * LDA + kk] = f2bf(w.w);
    }
    __syncthreads();

    FragBF af[2], bfr[4];
#pragma unroll
    for (int mt = 0; mt < 2; ++mt) {
      const bf16_t* p = sA + (wm * 32 + mt * 16 + l16) * LDA + hh * 8;
      af[mt].u[0] = *(const v4u*)p;
      af[mt].u[1] = *(const v4u*)(p + 16);
    }
#pragma unroll
    for (int nt = 0; nt < 4; ++nt) {
      const bf16_t* p = sB + (wn * 64 + nt * 16 + l16) * LDA + hh * 8;
      bfr[nt].u[0] = *(const v4u*)p;
      bfr[nt].u[1] = *(const v4u*)(p + 16);
    }
#pragma unroll
    for (int mt = 0; mt < 2; ++mt)
#pragma unroll
      for (int nt = 0; nt < 4; ++nt)
        acc[mt][nt] = wmma_bf16(af[mt].v, bfr[nt].v, acc[mt][nt]);
    __syncthreads();
  }

  // epilogue: BN1 per output channel, split into q/k/v (head-major bf16)
#pragma unroll
  for (int nt = 0; nt < 4; ++nt) {
    int col = col0 + wn * 64 + nt * 16 + l16;
    float sc = g1[col] * rsqrtf(v1[col] + 1e-5f);
    float sh = b1[col] - m1[col] * sc;
    int h = col >> 7, cc = col & 127;   // 128 channels per head: q32|k32|v64
#pragma unroll
    for (int mt = 0; mt < 2; ++mt) {
#pragma unroll
      for (int i = 0; i < 8; ++i) {
        int r = row0 + wm * 32 + mt * 16 + i + hh * 8;
        if (r < M) {
          int bb = r / S_, s = r - bb * S_;
          float y = acc[mt][nt][i] * sc + sh;
          size_t base = (size_t)(bb * HEADS_ + h) * S_ + s;
          if (cc < KD)          qb[base * KD + cc] = f2bf(y);
          else if (cc < 2 * KD) kb[base * KD + (cc - KD)] = f2bf(y);
          else                  vb[base * VD + (cc - 2 * KD)] = f2bf(y);
        }
      }
    }
  }
}

// ---------------------------------------------------------------------------
// Kernel 2: fused attention per (b,h): scores + bias, softmax, P@V, hard_swish
// Uses ~303KB dynamic LDS (CDNA5: up to 320KB per workgroup).
// q/k staged via CDNA5 async global->LDS DMA (ASYNCcnt), one-shot prologue.
// ---------------------------------------------------------------------------
__global__ __launch_bounds__(256) void attn_kernel(
    const bf16_t* __restrict__ qg, const bf16_t* __restrict__ kg,
    const bf16_t* __restrict__ vg, const float* __restrict__ biasx,
    bf16_t* __restrict__ outb) {
  extern __shared__ char smem[];
  bf16_t* sq = (bf16_t*)smem;          // SP x LDA
  bf16_t* sk = sq + SP * LDA;          // SP x LDA
  bf16_t* sv = sk + SP * LDA;          // vT: VD x LDV (K contiguous per column)
  bf16_t* sp = sv + VD * LDV;          // probs: SP x LDP bf16
  float*  ss = (float*)(sp + SP * LDP);// scores: S_ x LDSC f32

  const int tid = threadIdx.x, lane = tid & 31, wave = tid >> 5;
  const int l16 = lane & 15, hh = lane >> 4;
  const int b = blockIdx.x, h = blockIdx.y;
  const size_t qkbase = (size_t)(b * HEADS_ + h) * S_ * KD;
  const size_t vbase  = (size_t)(b * HEADS_ + h) * S_ * VD;

  // async-DMA q,k rows into LDS: 784 16-byte chunks each (row = 64B -> 4 chunks)
  const char* qgb = (const char*)(qg + qkbase);
  const char* kgb = (const char*)(kg + qkbase);
  for (int c = tid; c < (S_ * KD) / 8; c += 256) {
    int s = c >> 2, qq = c & 3;
    async_copy16(sq + s * LDA + qq * 8, qgb + c * 16);
    async_copy16(sk + s * LDA + qq * 8, kgb + c * 16);
  }
  for (int j = tid; j < (SP - S_) * LDA; j += 256) {   // zero pad rows 196..207
    sq[S_ * LDA + j] = f2bf(0.f);
    sk[S_ * LDA + j] = f2bf(0.f);
  }
  // load v transposed (scatter, stays on VGPR path): sv[d][s]
  for (int idx = tid; idx < S_ * VD; idx += 256) {
    int s = idx >> 6, d = idx & 63;
    sv[d * LDV + s] = vg[vbase + idx];
  }
  for (int j = tid; j < VD * (LDV - S_); j += 256) {   // zero K padding
    int d = j / (LDV - S_), s = S_ + j % (LDV - S_);
    sv[d * LDV + s] = f2bf(0.f);
  }
  async_wait0();
  __syncthreads();

  // Phase A: scores = q @ k^T * scale + bias  (13x13 tiles, 1 WMMA each, K=32)
  const float scale = 0.17677669529663687f;   // 32^-0.5
  for (int t = wave; t < 169; t += 8) {
    int qi = t / 13, ki = t - qi * 13;
    FragBF a, bb;
    const bf16_t* pa = sq + (qi * 16 + l16) * LDA + hh * 8;
    a.u[0] = *(const v4u*)pa; a.u[1] = *(const v4u*)(pa + 16);
    const bf16_t* pb = sk + (ki * 16 + l16) * LDA + hh * 8;
    bb.u[0] = *(const v4u*)pb; bb.u[1] = *(const v4u*)(pb + 16);
    v8f c = {};
    c = wmma_bf16(a.v, bb.v, c);
#pragma unroll
    for (int i = 0; i < 8; ++i) {
      int sqr = qi * 16 + i + hh * 8;
      int skc = ki * 16 + l16;
      if (sqr < S_ && skc < S_)
        ss[sqr * LDSC + skc] = c[i] * scale + biasx[((size_t)h * S_ + sqr) * S_ + skc];
    }
  }
  __syncthreads();

  // Phase B: row softmax, emit bf16 probs (padded K cols zeroed)
  if (tid < S_) {
    const float* row = ss + tid * LDSC;
    bf16_t* prow = sp + tid * LDP;
    float mx = -1e30f;
    for (int j = 0; j < S_; ++j) mx = fmaxf(mx, row[j]);
    float sum = 0.f;
    for (int j = 0; j < S_; ++j) {
      float e = __expf(row[j] - mx);
      sum += e;
      prow[j] = f2bf(e);
    }
    float inv = 1.f / sum;
    for (int j = 0; j < S_; ++j) prow[j] = f2bf((float)prow[j] * inv);
    for (int j = S_; j < LDP; ++j) prow[j] = f2bf(0.f);
  }
  __syncthreads();

  // Phase C: out = P @ V (13 q-tiles x 4 n-tiles, K = 224 in 7 WMMA steps),
  // fused hard_swish, bf16 store into [B*S, 768] for the proj GEMM.
  for (int t = wave; t < 52; t += 8) {
    int qi = t >> 2, nt = t & 3;
    v8f c = {};
#pragma unroll
    for (int kk = 0; kk < 7; ++kk) {
      FragBF a, bb;
      const bf16_t* pa = sp + (qi * 16 + l16) * LDP + kk * 32 + hh * 8;
      a.u[0] = *(const v4u*)pa; a.u[1] = *(const v4u*)(pa + 16);
      const bf16_t* pb = sv + (nt * 16 + l16) * LDV + kk * 32 + hh * 8;
      bb.u[0] = *(const v4u*)pb; bb.u[1] = *(const v4u*)(pb + 16);
      c = wmma_bf16(a.v, bb.v, c);
    }
#pragma unroll
    for (int i = 0; i < 8; ++i) {
      int s = qi * 16 + i + hh * 8;
      if (s < S_) {
        int d = nt * 16 + l16;
        float x = c[i];
        float hs = x * (fminf(fmaxf(x + 3.f, 0.f), 6.f) * (1.f / 6.f));
        outb[(size_t)(b * S_ + s) * PIN_ + h * VD + d] = f2bf(hs);
      }
    }
  }
}

// ---------------------------------------------------------------------------
// Kernel 3: out = hswish_out(bf16) @ W_proj, fused BN2, f32 result.
// A tile (bf16 in global) is staged via DOUBLE-BUFFERED async global->LDS DMA:
// tile t+1's DMA overlaps tile t's WMMAs; s_wait_asynccnt 2 lets the 2 newest
// per-thread async ops (next tile) stay in flight.
// ---------------------------------------------------------------------------
__global__ __launch_bounds__(256) void proj_gemm_kernel(
    const bf16_t* __restrict__ A, const float* __restrict__ W,
    const float* __restrict__ g2, const float* __restrict__ b2,
    const float* __restrict__ m2, const float* __restrict__ v2,
    float* __restrict__ out, int M) {
  __shared__ bf16_t sA[2][128 * LDA];
  __shared__ bf16_t sB[128 * LDA];
  const int tid = threadIdx.x;
  const int lane = tid & 31, wave = tid >> 5;
  const int wm = wave & 3, wn = wave >> 2;
  const int l16 = lane & 15, hh = lane >> 4;
  const int row0 = blockIdx.x * 128;
  const int col0 = blockIdx.y * 128;
  const int NT = PIN_ / 32;   // 24 K-tiles

  v8f zerov = {};
  v8f acc[2][4];
#pragma unroll
  for (int i = 0; i < 2; ++i)
#pragma unroll
    for (int j = 0; j < 4; ++j) acc[i][j] = zerov;

  auto issueA = [&](int buf, int k0) {
#pragma unroll
    for (int i = 0; i < 2; ++i) {
      int idx = tid + i * 256;
      int r = idx >> 2, c8 = idx & 3;
      int rr = row0 + r; if (rr >= M) rr = M - 1;
      async_copy16(&sA[buf][r * LDA + c8 * 8], A + (size_t)rr * PIN_ + k0 + c8 * 8);
    }
  };

  issueA(0, 0);   // prologue DMA for tile 0

#pragma unroll
  for (int t = 0; t < NT; ++t) {
    const int k0 = t * 32;
    if (t + 1 < NT) {
      issueA((t + 1) & 1, k0 + 32);   // overlap next tile's DMA with this tile
      __builtin_prefetch(W + (size_t)(k0 + 32 + (tid & 31)) * D_ + col0, 0, 3);
    }
    // B tile: 32x128 f32 -> bf16, transposed (conversion keeps VGPR path)
#pragma unroll
    for (int i = 0; i < 4; ++i) {
      int idx = tid + i * 256;
      int kk = idx >> 5, c4 = idx & 31;
      const float4 w = *(const float4*)(W + (size_t)(k0 + kk) * D_ + col0 + c4 * 4);
      int cb = c4 * 4;
      sB[(cb + 0) * LDA + kk] = f2bf(w.x);
      sB[(cb + 1) * LDA + kk] = f2bf(w.y);
      sB[(cb + 2) * LDA + kk] = f2bf(w.z);
      sB[(cb + 3) * LDA + kk] = f2bf(w.w);
    }
    if (t + 1 < NT) async_wait2(); else async_wait0();  // tile t's DMA landed
    __syncthreads();

    const bf16_t* sAt = sA[t & 1];
    FragBF af[2], bfr[4];
#pragma unroll
    for (int mt = 0; mt < 2; ++mt) {
      const bf16_t* p = sAt + (wm * 32 + mt * 16 + l16) * LDA + hh * 8;
      af[mt].u[0] = *(const v4u*)p;
      af[mt].u[1] = *(const v4u*)(p + 16);
    }
#pragma unroll
    for (int nt = 0; nt < 4; ++nt) {
      const bf16_t* p = sB + (wn * 64 + nt * 16 + l16) * LDA + hh * 8;
      bfr[nt].u[0] = *(const v4u*)p;
      bfr[nt].u[1] = *(const v4u*)(p + 16);
    }
#pragma unroll
    for (int mt = 0; mt < 2; ++mt)
#pragma unroll
      for (int nt = 0; nt < 4; ++nt)
        acc[mt][nt] = wmma_bf16(af[mt].v, bfr[nt].v, acc[mt][nt]);
    __syncthreads();   // all waves done with sB and sA[t&1] before reuse
  }

#pragma unroll
  for (int nt = 0; nt < 4; ++nt) {
    int col = col0 + wn * 64 + nt * 16 + l16;
    float sc = g2[col] * rsqrtf(v2[col] + 1e-5f);
    float sh = b2[col] - m2[col] * sc;
#pragma unroll
    for (int mt = 0; mt < 2; ++mt) {
#pragma unroll
      for (int i = 0; i < 8; ++i) {
        int r = row0 + wm * 32 + mt * 16 + i + hh * 8;
        if (r < M) out[(size_t)r * D_ + col] = acc[mt][nt][i] * sc + sh;
      }
    }
  }
}

// ---------------------------------------------------------------------------
extern "C" void kernel_launch(void* const* d_in, const int* in_sizes, int n_in,
                              void* d_out, int out_size, void* d_ws, size_t ws_size,
                              hipStream_t stream) {
  (void)n_in; (void)out_size; (void)ws_size;
  const float* X   = (const float*)d_in[0];
  const float* Wq  = (const float*)d_in[1];
  const float* g1  = (const float*)d_in[2];
  const float* b1  = (const float*)d_in[3];
  const float* m1  = (const float*)d_in[4];
  const float* v1  = (const float*)d_in[5];
  const float* Wp  = (const float*)d_in[6];
  const float* g2  = (const float*)d_in[7];
  const float* b2  = (const float*)d_in[8];
  const float* m2  = (const float*)d_in[9];
  const float* v2  = (const float*)d_in[10];
  const float* ab  = (const float*)d_in[11];
  const int*  idxs = (const int*)d_in[12];

  const int B = in_sizes[0] / (S_ * D_);
  const int M = B * S_;
  const int noff = in_sizes[11] / HEADS_;

  char* ws = (char*)d_ws;
  size_t off = 0;
  auto carve = [&](size_t bytes) -> char* {
    char* p = ws + off;
    off += (bytes + 255) & ~(size_t)255;
    return p;
  };
  bf16_t* qb     = (bf16_t*)carve((size_t)M * HEADS_ * KD * 2);
  bf16_t* kb     = (bf16_t*)carve((size_t)M * HEADS_ * KD * 2);
  bf16_t* vb     = (bf16_t*)carve((size_t)M * HEADS_ * VD * 2);
  float*  biasx  = (float*) carve((size_t)HEADS_ * S_ * S_ * 4);
  bf16_t* outmid = (bf16_t*)carve((size_t)M * PIN_ * 2);

  const int tb = HEADS_ * S_ * S_;
  bias_expand_kernel<<<(tb + 255) / 256, 256, 0, stream>>>(ab, idxs, biasx, noff);

  qkv_gemm_kernel<<<dim3((M + 127) / 128, QKV_ / 128), 256, 0, stream>>>(
      X, Wq, g1, b1, m1, v1, qb, kb, vb, M);

  const size_t smem2 =
      (size_t)(2 * SP * LDA + VD * LDV + SP * LDP) * sizeof(bf16_t) +
      (size_t)S_ * LDSC * sizeof(float);   // 310,608 B < 320KB WGP LDS
  attn_kernel<<<dim3(B, HEADS_), 256, smem2, stream>>>(qb, kb, vb, biasx, outmid);

  proj_gemm_kernel<<<dim3((M + 127) / 128, D_ / 128), 256, 0, stream>>>(
      outmid, Wp, g2, b2, m2, v2, (float*)d_out, M);
}
